// VectorQuantizerEMA_14302241096429
// MI455X (gfx1250) — matrix operand order repaired and checked by reading, not compiled, not run on hardware
//
#include <hip/hip_runtime.h>
#include <hip/hip_bf16.h>

typedef unsigned short ushort_t;
typedef __attribute__((ext_vector_type(16))) __bf16 v16bf;
typedef __attribute__((ext_vector_type(8)))  float  v8f;

#define N_TOKENS 131072
#define N_CODES  1024
#define D_CODE   256
#define DECAY    0.97f
#define ONE_M_D  (1.0f - 0.97f)
#define BETA     0.25f
#define EPSF     1e-5f

union Frag {
    v16bf v;
    uint4 q[2];
};

__device__ __forceinline__ ushort_t f2bf(float f) {
    unsigned u = __float_as_uint(f);
    unsigned r = u + 0x7fffu + ((u >> 16) & 1u);   // round-to-nearest-even
    return (ushort_t)(r >> 16);
}

__device__ __forceinline__ float waveRedSum(float v) {
    #pragma unroll
    for (int m = 16; m >= 1; m >>= 1) v += __shfl_xor(v, m, 32);
    return v;
}

// ---------------------------------------------------------------------------
// K1: normalize codebook rows -> bf16  (wave per row)
// ---------------------------------------------------------------------------
__global__ __launch_bounds__(256) void k_norm_cb(const float* __restrict__ cb,
                                                 ushort_t* __restrict__ cbb) {
    int wave = threadIdx.x >> 5, lane = threadIdx.x & 31;
    int row  = blockIdx.x * 8 + wave;                 // grid=128 -> 1024 rows
    const float4* src = (const float4*)(cb + (size_t)row * D_CODE);
    float4 x0 = src[lane * 2], x1 = src[lane * 2 + 1];
    float ss = x0.x*x0.x + x0.y*x0.y + x0.z*x0.z + x0.w*x0.w
             + x1.x*x1.x + x1.y*x1.y + x1.z*x1.z + x1.w*x1.w;
    ss = waveRedSum(ss);
    float rn = 1.0f / fmaxf(sqrtf(ss), 1e-12f);
    alignas(16) ushort_t o[8] = {
        f2bf(x0.x*rn), f2bf(x0.y*rn), f2bf(x0.z*rn), f2bf(x0.w*rn),
        f2bf(x1.x*rn), f2bf(x1.y*rn), f2bf(x1.z*rn), f2bf(x1.w*rn)};
    *(uint4*)(cbb + (size_t)row * D_CODE + lane * 8) = *(const uint4*)o;
}

// ---------------------------------------------------------------------------
// K2: fused normalize(z) -> bf16 LDS, sim = z*cb^T via WMMA, row-argmax.
//     block = 64 threads (2 waves), 64 tokens/block.
//     Each wave owns 32 tokens = TWO 16x16 M-tiles sharing every B fragment
//     (2 WMMAs per B load pair), with a 2-deep rotating B prefetch buffer.
// ---------------------------------------------------------------------------
#define ZPAD 264   // 264 ushorts = 528B row stride -> conflict-free b128 reads

__global__ __launch_bounds__(64) void k_gemm_argmax(
        const float* __restrict__ z_e, const ushort_t* __restrict__ cbb,
        float* __restrict__ out_codes_f, int* __restrict__ codes_i,
        float* __restrict__ rnorm_ws) {
    __shared__ ushort_t zA[64][ZPAD];
    int wave = threadIdx.x >> 5, lane = threadIdx.x & 31;
    int rowBase = wave * 32;                 // wave-private rows [rowBase, +32)
    int tokBase = blockIdx.x * 64;

    // ---- normalize 32 rows of z_e into bf16 LDS (wave-private rows) ----
    #pragma unroll 4
    for (int r = 0; r < 32; ++r) {
        int row = rowBase + r;
        int tok = tokBase + row;
        const float4* src = (const float4*)(z_e + (size_t)tok * D_CODE);
        float4 x0 = src[lane * 2], x1 = src[lane * 2 + 1];
        float ss = x0.x*x0.x + x0.y*x0.y + x0.z*x0.z + x0.w*x0.w
                 + x1.x*x1.x + x1.y*x1.y + x1.z*x1.z + x1.w*x1.w;
        ss = waveRedSum(ss);
        float rn = 1.0f / fmaxf(sqrtf(ss), 1e-12f);
        alignas(16) ushort_t o[8] = {
            f2bf(x0.x*rn), f2bf(x0.y*rn), f2bf(x0.z*rn), f2bf(x0.w*rn),
            f2bf(x1.x*rn), f2bf(x1.y*rn), f2bf(x1.z*rn), f2bf(x1.w*rn)};
        *(uint4*)&zA[row][lane * 8] = *(const uint4*)o;
        if (lane == 0) rnorm_ws[tok] = rn;
    }
    __syncthreads();

    // ---- hoist all A fragments for both M-tiles (K = 256) ----
    // A 16x32 bf16 layout: lane%16 = M row; lanes 0-15: K=0..15,
    // lanes 16-31: K=16..31 (contiguous, matches B)
    Frag A0[8], A1[8];
    int arow0  = rowBase + (lane & 15);
    int arow1  = arow0 + 16;
    int abase0 = (lane < 16) ? 0 : 16;
    #pragma unroll
    for (int ks = 0; ks < 8; ++ks) {
        int base = ks * 32 + abase0;
        A0[ks].q[0] = *(const uint4*)&zA[arow0][base];
        A0[ks].q[1] = *(const uint4*)&zA[arow0][base + 8];
        A1[ks].q[0] = *(const uint4*)&zA[arow1][base];
        A1[ks].q[1] = *(const uint4*)&zA[arow1][base + 8];
    }

    float bestv[2][8];
    int   besti[2][8];
    #pragma unroll
    for (int t = 0; t < 2; ++t)
        #pragma unroll
        for (int i = 0; i < 8; ++i) { bestv[t][i] = -3.402823466e38f; besti[t][i] = 0; }

    // B 32x16 bf16 layout: lane%16 = N col; lanes 0-15 hold K=0..15,
    // lanes 16-31 hold K=16..31 (contiguous) -> one 32B row-major load.
    int bcol  = lane & 15;
    int bkoff = (lane < 16) ? 0 : 16;
    // addr(ct, ks) = bptr + ct*4096 + ks*32   (ushort units)
    const ushort_t* bptr = cbb + (size_t)bcol * D_CODE + bkoff;

    Frag Bb[2];
    {   // prologue: prefetch fragment (ct=0, ks=0)
        const uint4* q = (const uint4*)bptr;
        Bb[0].q[0] = q[0];
        Bb[0].q[1] = q[1];
    }

    for (int ct = 0; ct < 64; ++ct) {
        const ushort_t* brow = bptr + (size_t)ct * 4096;
        v8f acc0 = {0.f,0.f,0.f,0.f,0.f,0.f,0.f,0.f};
        v8f acc1 = {0.f,0.f,0.f,0.f,0.f,0.f,0.f,0.f};
        #pragma unroll
        for (int ks = 0; ks < 8; ++ks) {
            int cur = ks & 1, nxt = cur ^ 1;
            // prefetch next fragment: (ct, ks+1), or (ct+1, 0) after ks=7.
            // At ct=63/ks=7 this over-reads 32B into the next ws region
            // (in-bounds allocation, value never consumed).
            const uint4* q = (const uint4*)(brow + (ks < 7 ? (ks + 1) * 32 : 4096));
            Bb[nxt].q[0] = q[0];
            Bb[nxt].q[1] = q[1];
            acc0 = __builtin_amdgcn_wmma_f32_16x16x32_bf16(
                       false, A0[ks].v, false, Bb[cur].v, (short)0, acc0, false, false);
            acc1 = __builtin_amdgcn_wmma_f32_16x16x32_bf16(
                       false, A1[ks].v, false, Bb[cur].v, (short)0, acc1, false, false);
        }
        int col = ct * 16 + bcol;
        #pragma unroll
        for (int i = 0; i < 8; ++i) {
            if (acc0[i] > bestv[0][i]) { bestv[0][i] = acc0[i]; besti[0][i] = col; }
            if (acc1[i] > bestv[1][i]) { bestv[1][i] = acc1[i]; besti[1][i] = col; }
        }
    }

    // ---- argmax across the 16 lanes sharing a row (first-max tiebreak) ----
    #pragma unroll
    for (int m = 1; m < 16; m <<= 1) {
        #pragma unroll
        for (int t = 0; t < 2; ++t) {
            #pragma unroll
            for (int i = 0; i < 8; ++i) {
                float ov = __shfl_xor(bestv[t][i], m, 32);
                int   oi = __shfl_xor(besti[t][i], m, 32);
                if (ov > bestv[t][i] || (ov == bestv[t][i] && oi < besti[t][i])) {
                    bestv[t][i] = ov; besti[t][i] = oi;
                }
            }
        }
    }
    if ((lane & 15) == 0) {
        int half = (lane >= 16) ? 8 : 0;            // acc VGPR i -> row half+i
        #pragma unroll
        for (int t = 0; t < 2; ++t) {
            int rbase = rowBase + t * 16 + half;
            #pragma unroll
            for (int i = 0; i < 8; ++i) {
                int tok = tokBase + rbase + i;
                out_codes_f[tok] = (float)besti[t][i];
                codes_i[tok]     = besti[t][i];
            }
        }
    }
}

// ---------------------------------------------------------------------------
// K3: counts histogram + dw segment-sum scatter (wave per token)
// ---------------------------------------------------------------------------
__global__ __launch_bounds__(256) void k_scatter(
        const float* __restrict__ z_e, const int* __restrict__ codes_i,
        const float* __restrict__ rnorm_ws, float* __restrict__ dw,
        float* __restrict__ counts) {
    int wave = threadIdx.x >> 5, lane = threadIdx.x & 31;
    int tok = blockIdx.x * 8 + wave;
    int c = codes_i[tok];
    float rn = rnorm_ws[tok];
    const float4* src = (const float4*)(z_e + (size_t)tok * D_CODE);
    float4 x0 = src[lane * 2], x1 = src[lane * 2 + 1];
    float* dst = dw + (size_t)c * D_CODE + lane * 8;
    unsafeAtomicAdd(dst + 0, x0.x * rn);
    unsafeAtomicAdd(dst + 1, x0.y * rn);
    unsafeAtomicAdd(dst + 2, x0.z * rn);
    unsafeAtomicAdd(dst + 3, x0.w * rn);
    unsafeAtomicAdd(dst + 4, x1.x * rn);
    unsafeAtomicAdd(dst + 5, x1.y * rn);
    unsafeAtomicAdd(dst + 6, x1.z * rn);
    unsafeAtomicAdd(dst + 7, x1.w * rn);
    if (lane == 0) unsafeAtomicAdd(counts + c, 1.0f);
}

// ---------------------------------------------------------------------------
// K4a: new_cs, n = sum(new_cs), cluster_size (single block of 1024)
// ---------------------------------------------------------------------------
__global__ __launch_bounds__(1024) void k_ema_cs(
        const float* __restrict__ ema_cs, const float* __restrict__ counts,
        float* __restrict__ cluster_size) {
    __shared__ float red[32];
    int tid = threadIdx.x;
    float ncs = DECAY * ema_cs[tid] + ONE_M_D * counts[tid];
    float s = waveRedSum(ncs);
    if ((tid & 31) == 0) red[tid >> 5] = s;
    __syncthreads();
    if (tid < 32) { float v = red[tid]; v = waveRedSum(v); if (tid == 0) red[0] = v; }
    __syncthreads();
    float n = red[0];
    cluster_size[tid] = (ncs + EPSF) / (n + (float)N_CODES * EPSF) * n;
}

// ---------------------------------------------------------------------------
// K4b: embed = l2norm((decay*ema_w + (1-decay)*dw) / cluster_size)
// ---------------------------------------------------------------------------
__global__ __launch_bounds__(256) void k_embed(
        const float* __restrict__ ema_w, const float* __restrict__ dw,
        const float* __restrict__ cluster_size, float* __restrict__ embed) {
    int wave = threadIdx.x >> 5, lane = threadIdx.x & 31;
    int row = blockIdx.x * 8 + wave;
    float inv_cs = 1.0f / cluster_size[row];
    const float4* ew = (const float4*)(ema_w + (size_t)row * D_CODE);
    const float4* dq = (const float4*)(dw    + (size_t)row * D_CODE);
    float4 e0 = ew[lane*2], e1 = ew[lane*2+1];
    float4 d0 = dq[lane*2], d1 = dq[lane*2+1];
    float x[8] = {
        (DECAY*e0.x + ONE_M_D*d0.x) * inv_cs, (DECAY*e0.y + ONE_M_D*d0.y) * inv_cs,
        (DECAY*e0.z + ONE_M_D*d0.z) * inv_cs, (DECAY*e0.w + ONE_M_D*d0.w) * inv_cs,
        (DECAY*e1.x + ONE_M_D*d1.x) * inv_cs, (DECAY*e1.y + ONE_M_D*d1.y) * inv_cs,
        (DECAY*e1.z + ONE_M_D*d1.z) * inv_cs, (DECAY*e1.w + ONE_M_D*d1.w) * inv_cs};
    float ss = 0.f;
    #pragma unroll
    for (int i = 0; i < 8; ++i) ss += x[i] * x[i];
    ss = waveRedSum(ss);
    float rn = 1.0f / fmaxf(sqrtf(ss), 1e-12f);
    float4 o0 = {x[0]*rn, x[1]*rn, x[2]*rn, x[3]*rn};
    float4 o1 = {x[4]*rn, x[5]*rn, x[6]*rn, x[7]*rn};
    float4* out = (float4*)(embed + (size_t)row * D_CODE);
    out[lane*2] = o0; out[lane*2+1] = o1;
}

// ---------------------------------------------------------------------------
// K5: z_q_out = embed[codes]  (embed is unit-norm already) + MSE loss
// ---------------------------------------------------------------------------
__global__ __launch_bounds__(256) void k_gather_loss(
        const float* __restrict__ z_e, const int* __restrict__ codes_i,
        const float* __restrict__ embed, float* __restrict__ out_zq,
        float* __restrict__ loss_slot) {
    __shared__ float red[8];
    int wave = threadIdx.x >> 5, lane = threadIdx.x & 31;
    int tok = blockIdx.x * 8 + wave;
    int c = codes_i[tok];
    const float4* eq = (const float4*)(embed + (size_t)c * D_CODE);
    const float4* ze = (const float4*)(z_e + (size_t)tok * D_CODE);
    float4 q0 = eq[lane*2], q1 = eq[lane*2+1];
    float4 z0 = ze[lane*2], z1 = ze[lane*2+1];
    float4* out = (float4*)(out_zq + (size_t)tok * D_CODE);
    out[lane*2] = q0; out[lane*2+1] = q1;
    float dx;
    float s = 0.f;
    dx = z0.x-q0.x; s += dx*dx;  dx = z0.y-q0.y; s += dx*dx;
    dx = z0.z-q0.z; s += dx*dx;  dx = z0.w-q0.w; s += dx*dx;
    dx = z1.x-q1.x; s += dx*dx;  dx = z1.y-q1.y; s += dx*dx;
    dx = z1.z-q1.z; s += dx*dx;  dx = z1.w-q1.w; s += dx*dx;
    s = waveRedSum(s);
    if (lane == 0) red[wave] = s;
    __syncthreads();
    if (threadIdx.x == 0) {
        float t = 0.f;
        #pragma unroll
        for (int i = 0; i < 8; ++i) t += red[i];
        unsafeAtomicAdd(loss_slot, t * (BETA / ((float)N_TOKENS * (float)D_CODE)));
    }
}

// ---------------------------------------------------------------------------
extern "C" void kernel_launch(void* const* d_in, const int* in_sizes, int n_in,
                              void* d_out, int out_size, void* d_ws, size_t ws_size,
                              hipStream_t stream) {
    const float* z_e    = (const float*)d_in[0];
    const float* cb     = (const float*)d_in[1];
    const float* ema_cs = (const float*)d_in[2];
    const float* ema_w  = (const float*)d_in[3];

    char* ws = (char*)d_ws;
    ushort_t* cbb        = (ushort_t*)(ws);                      // 1024*256*2 = 512K
    int*      codes_i    = (int*)     (ws + 524288);             // 131072*4  = 512K
    float*    rnorm_ws   = (float*)   (ws + 1048576);            // 131072*4  = 512K
    float*    counts     = (float*)   (ws + 1572864);            // 1024*4
    float*    cluster_sz = (float*)   (ws + 1576960);            // 1024*4
    float*    dw         = (float*)   (ws + 1581056);            // 1024*256*4 = 1M
    float*    embed      = (float*)   (ws + 2629632);            // 1024*256*4 = 1M

    float* out_zq      = (float*)d_out;                                   // [N,256]
    float* out_codes_f = (float*)d_out + (size_t)N_TOKENS * D_CODE;       // [N]
    float* loss_slot   = out_codes_f + N_TOKENS;                          // [1]

    hipMemsetAsync(counts, 0, N_CODES * sizeof(float), stream);
    hipMemsetAsync(dw, 0, (size_t)N_CODES * D_CODE * sizeof(float), stream);
    hipMemsetAsync(loss_slot, 0, sizeof(float), stream);

    k_norm_cb    <<<N_CODES / 8, 256, 0, stream>>>(cb, cbb);
    k_gemm_argmax<<<N_TOKENS / 64, 64, 0, stream>>>(z_e, cbb, out_codes_f,
                                                    codes_i, rnorm_ws);
    k_scatter    <<<N_TOKENS / 8, 256, 0, stream>>>(z_e, codes_i, rnorm_ws,
                                                    dw, counts);
    k_ema_cs     <<<1, 1024, 0, stream>>>(ema_cs, counts, cluster_sz);
    k_embed      <<<N_CODES / 8, 256, 0, stream>>>(ema_w, dw, cluster_sz, embed);
    k_gather_loss<<<N_TOKENS / 8, 256, 0, stream>>>(z_e, codes_i, embed,
                                                    out_zq, loss_slot);
}
